// MambaBlock_8358006358190
// MI455X (gfx1250) — compile-verified, matrix operand order checked by reading, not run on hardware
//
#include <hip/hip_runtime.h>

// ---------------- CDNA5 WMMA types ----------------
typedef __attribute__((ext_vector_type(16))) __bf16 v16bf;
typedef __attribute__((ext_vector_type(8)))  float  v8f;
typedef unsigned int u32;
typedef __attribute__((ext_vector_type(4)))  u32    u32x4;
typedef __attribute__((ext_vector_type(4)))  int    iv4;
typedef unsigned short ushort_t;

union FragAB { v16bf v; u32x4 q[2]; };
union FragC  { v8f   v; float f[8]; };

// gfx1250 async global->LDS copy path (ASYNCcnt-tracked), guarded so the
// build cannot regress if the toolchain lacks the builtins.
// Probe round 2 showed the builtin's params are int4-in-addrspace pointers:
//   (int4 __device__* /*as1*/, int4 __shared__* /*as3*/, imm offset, imm cpol)
#ifndef __has_builtin
#define __has_builtin(x) 0
#endif
#if __has_builtin(__builtin_amdgcn_global_load_async_to_lds_b128) && \
    __has_builtin(__builtin_amdgcn_s_wait_asynccnt)
#define USE_ASYNC_COPY 1
typedef __attribute__((address_space(1))) iv4 iv4_g;
typedef __attribute__((address_space(3))) iv4 iv4_l;
#else
#define USE_ASYNC_COPY 0
#endif

// ---------------- constants ----------------
#define D_MODEL 1024
#define D_STATE 16
#define D_CONV  4
#define D_INNER 2048
#define DT_RANK 64
#define BATCH   2
#define SEQ     2048
#define ROWS    (BATCH * SEQ)   // 4096

__device__ __forceinline__ ushort_t f2bf(float f) {
    u32 u = __float_as_uint(f);
    u32 r = u + 0x7FFFu + ((u >> 16) & 1u);   // round to nearest even
    return (ushort_t)(r >> 16);
}

// ---------------- fp32 -> bf16 convert ----------------
__global__ __launch_bounds__(256)
void f2bf_kernel(const float* __restrict__ s, ushort_t* __restrict__ d, int n) {
    int i = blockIdx.x * 256 + threadIdx.x;
    if (i < n) d[i] = f2bf(s[i]);
}

// ---------------- RMSNorm -> bf16 ----------------
__global__ __launch_bounds__(256)
void rmsnorm_kernel(const float* __restrict__ x, const float* __restrict__ nw,
                    ushort_t* __restrict__ xn) {
    int row = blockIdx.x;
    const float* xr = x + (size_t)row * D_MODEL;
    __shared__ float red[256];
    float s = 0.f;
#pragma unroll
    for (int i = 0; i < 4; ++i) { float v = xr[threadIdx.x + i * 256]; s += v * v; }
    red[threadIdx.x] = s; __syncthreads();
    for (int off = 128; off > 0; off >>= 1) {
        if (threadIdx.x < off) red[threadIdx.x] += red[threadIdx.x + off];
        __syncthreads();
    }
    float scale = rsqrtf(red[0] * (1.f / D_MODEL) + 1e-6f);
    ushort_t* o = xn + (size_t)row * D_MODEL;
#pragma unroll
    for (int i = 0; i < 4; ++i) {
        int c = threadIdx.x + i * 256;
        o[c] = f2bf(xr[c] * scale * nw[c]);
    }
}

// ---------------- generic bf16 WMMA GEMM:  C[m,n] = sum_k A[m,k]*W[n,k] (+Res) ----
// A: [M,K] bf16 row-major, W: [N,K] bf16 row-major, C: fp32 with leading dim ldc.
// Block = 128 threads (4 waves), 64x64 tile, BK=64 (two 16x16x32 wmma K-steps).
// Staging uses gfx1250 GLOBAL_LOAD_ASYNC_TO_LDS_B128 when available.
#define BM 64
#define BN 64
#define BK 64

__global__ __launch_bounds__(128)
void gemm_bf16_wmma(const ushort_t* __restrict__ A, const ushort_t* __restrict__ W,
                    float* __restrict__ C, const float* __restrict__ Res,
                    int M, int N, int K, int ldc) {
    __shared__ ushort_t As[BM][BK];
    __shared__ ushort_t Bs[BN][BK];

    const int tid   = threadIdx.x;
    const int lane  = tid & 31;
    const int wave  = tid >> 5;
    const int waveM = wave >> 1;       // 2x2 waves over 64x64
    const int waveN = wave & 1;
    const int m0 = blockIdx.y * BM;
    const int n0 = blockIdx.x * BN;

    FragC c[2][2];
#pragma unroll
    for (int mt = 0; mt < 2; ++mt)
#pragma unroll
        for (int nt = 0; nt < 2; ++nt)
#pragma unroll
            for (int r = 0; r < 8; ++r) c[mt][nt].f[r] = 0.f;

    // staging: each of 128 threads copies half a 64-element row (32 bf16 = 4x16B)
    const int lr = tid >> 1;
    const int lc = (tid & 1) * 32;
    int gnB = n0 + lr; if (gnB > N - 1) gnB = N - 1;   // clamp (N=96 case)

    for (int k0 = 0; k0 < K; k0 += BK) {
        const ushort_t* gA = A + (size_t)(m0 + lr) * K + k0 + lc;
        const ushort_t* gB = W + (size_t)gnB * K + k0 + lc;
#if USE_ASYNC_COPY
        {
            iv4_g* ga = (iv4_g*)(void*)gA;
            iv4_l* la = (iv4_l*)&As[lr][lc];
            __builtin_amdgcn_global_load_async_to_lds_b128(ga, la, 0, 0);
            __builtin_amdgcn_global_load_async_to_lds_b128(ga + 1, la + 1, 0, 0);
            __builtin_amdgcn_global_load_async_to_lds_b128(ga + 2, la + 2, 0, 0);
            __builtin_amdgcn_global_load_async_to_lds_b128(ga + 3, la + 3, 0, 0);
            iv4_g* gb = (iv4_g*)(void*)gB;
            iv4_l* lb = (iv4_l*)&Bs[lr][lc];
            __builtin_amdgcn_global_load_async_to_lds_b128(gb, lb, 0, 0);
            __builtin_amdgcn_global_load_async_to_lds_b128(gb + 1, lb + 1, 0, 0);
            __builtin_amdgcn_global_load_async_to_lds_b128(gb + 2, lb + 2, 0, 0);
            __builtin_amdgcn_global_load_async_to_lds_b128(gb + 3, lb + 3, 0, 0);
        }
#else
        {
            const u32x4* src = (const u32x4*)gA;
            u32x4* dst = (u32x4*)&As[lr][lc];
            dst[0] = src[0]; dst[1] = src[1]; dst[2] = src[2]; dst[3] = src[3];
        }
        {
            const u32x4* src = (const u32x4*)gB;
            u32x4* dst = (u32x4*)&Bs[lr][lc];
            dst[0] = src[0]; dst[1] = src[1]; dst[2] = src[2]; dst[3] = src[3];
        }
#endif
        if (k0 + BK < K) {   // gfx1250 global_prefetch_b8 for next K-tile
            __builtin_prefetch(gA + BK, 0, 1);
            __builtin_prefetch(gB + BK, 0, 1);
        }
#if USE_ASYNC_COPY
        __builtin_amdgcn_s_wait_asynccnt(0);
#endif
        __syncthreads();

        const int fr = lane & 15;          // row/col within 16-tile
        const int kg = (lane >> 4) * 8;    // K half-group select (ISA 7.12.2)
#pragma unroll
        for (int kk = 0; kk < BK; kk += 32) {
            FragAB a[2], b[2];
#pragma unroll
            for (int mt = 0; mt < 2; ++mt) {
                const ushort_t* p = &As[waveM * 32 + mt * 16 + fr][kk + kg];
                a[mt].q[0] = *(const u32x4*)p;
                a[mt].q[1] = *(const u32x4*)(p + 16);
            }
#pragma unroll
            for (int nt = 0; nt < 2; ++nt) {
                const ushort_t* p = &Bs[waveN * 32 + nt * 16 + fr][kk + kg];
                b[nt].q[0] = *(const u32x4*)p;
                b[nt].q[1] = *(const u32x4*)(p + 16);
            }
#pragma unroll
            for (int mt = 0; mt < 2; ++mt)
#pragma unroll
                for (int nt = 0; nt < 2; ++nt)
                    c[mt][nt].v = __builtin_amdgcn_wmma_f32_16x16x32_bf16(
                        false, a[mt].v, false, b[nt].v,
                        (short)0, c[mt][nt].v, false, false);
        }
        __syncthreads();
    }

    // store D: VGPR r -> row r + 8*(lane>=16), lane%16 -> column
    const int fr = lane & 15;
    const int rsel = (lane >> 4) * 8;
#pragma unroll
    for (int mt = 0; mt < 2; ++mt) {
#pragma unroll
        for (int nt = 0; nt < 2; ++nt) {
            int n = n0 + waveN * 32 + nt * 16 + fr;
            if (n < N) {
                int mb = m0 + waveM * 32 + mt * 16 + rsel;
#pragma unroll
                for (int r = 0; r < 8; ++r) {
                    size_t idx = (size_t)(mb + r) * ldc + n;
                    float v = c[mt][nt].f[r];
                    if (Res) v += Res[idx];
                    C[idx] = v;
                }
            }
        }
    }
}

// ---------------- depthwise causal conv (k=4) + SiLU ----------------
__global__ __launch_bounds__(256)
void conv_silu_kernel(const float* __restrict__ xz, const float* __restrict__ cw,
                      const float* __restrict__ cb, float* __restrict__ x_conv,
                      ushort_t* __restrict__ x_conv_bf) {
    size_t i = (size_t)blockIdx.x * 256 + threadIdx.x;   // over ROWS*D_INNER
    int c = (int)(i & (D_INNER - 1));
    size_t row = i >> 11;
    int t = (int)(row & (SEQ - 1));
    size_t b = row >> 11;
    float acc = cb[c];
#pragma unroll
    for (int j = 0; j < D_CONV; ++j) {
        int tt = t - (D_CONV - 1) + j;
        if (tt >= 0)
            acc += cw[c * D_CONV + j] * xz[((b << 11) + tt) * (2 * D_INNER) + c];
    }
    float v = acc / (1.f + __expf(-acc));   // silu
    x_conv[i] = v;
    x_conv_bf[i] = f2bf(v);
}

// ---------------- dt_raw (cols 0..63 of x_dbc) -> bf16 ----------------
__global__ __launch_bounds__(256)
void dtraw_bf_kernel(const float* __restrict__ x_dbc, ushort_t* __restrict__ dtb) {
    int i = blockIdx.x * 256 + threadIdx.x;              // over ROWS*DT_RANK
    int r = i & (DT_RANK - 1);
    int m = i >> 6;
    dtb[i] = f2bf(x_dbc[(size_t)m * 96 + r]);
}

// ---------------- softplus(+bias), clip, du = dc * x_conv ----------------
__global__ __launch_bounds__(256)
void dt_epilogue_kernel(float* __restrict__ dc /* in: pre, out: dc */,
                        const float* __restrict__ bias,
                        const float* __restrict__ x_conv,
                        float* __restrict__ du) {
    size_t i = (size_t)blockIdx.x * 256 + threadIdx.x;
    int c = (int)(i & (D_INNER - 1));
    float v = dc[i] + bias[c];
    float sp = (v > 20.f) ? v : log1pf(__expf(v));
    float d = fminf(fmaxf(sp, -10.f), 1.f);
    dc[i] = d;
    du[i] = d * x_conv[i];
}

// ---------------- sequential SSM scan + gating -> bf16 ----------------
__global__ __launch_bounds__(256)
void scan_kernel(const float* __restrict__ dc, const float* __restrict__ du,
                 const float* __restrict__ x_dbc, const float* __restrict__ x_conv,
                 const float* __restrict__ xz, const float* __restrict__ A_log,
                 const float* __restrict__ Dp, ushort_t* __restrict__ yg) {
    int g = blockIdx.x * 256 + threadIdx.x;   // 0..4095 -> (b, channel)
    int b = g >> 11;
    int c = g & (D_INNER - 1);
    float a_s[D_STATE], h[D_STATE];
#pragma unroll
    for (int s = 0; s < D_STATE; ++s) {
        a_s[s] = -__expf(A_log[c * D_STATE + s]);
        h[s] = 0.f;
    }
    float Dc = Dp[c];
    for (int t = 0; t < SEQ; ++t) {
        size_t row = (size_t)b * SEQ + t;
        float dcv = dc[row * D_INNER + c];
        float duv = du[row * D_INNER + c];
        const float* bc = x_dbc + row * 96;
        float y = 0.f;
#pragma unroll
        for (int s = 0; s < D_STATE; ++s) {
            h[s] = __expf(dcv * a_s[s]) * h[s] + duv * bc[DT_RANK + s];
            y += h[s] * bc[DT_RANK + D_STATE + s];
        }
        float xc = x_conv[row * D_INNER + c];
        float zv = xz[row * (2 * D_INNER) + D_INNER + c];
        float sz = zv / (1.f + __expf(-zv));
        y = (y + xc * Dc) * sz;
        yg[row * D_INNER + c] = f2bf(y);
    }
}

// ---------------- launch ----------------
extern "C" void kernel_launch(void* const* d_in, const int* in_sizes, int n_in,
                              void* d_out, int out_size, void* d_ws, size_t ws_size,
                              hipStream_t stream) {
    const float* x        = (const float*)d_in[0];
    const float* in_w     = (const float*)d_in[1];
    const float* conv_w   = (const float*)d_in[2];
    const float* conv_b   = (const float*)d_in[3];
    const float* xproj_w  = (const float*)d_in[4];
    const float* dtproj_w = (const float*)d_in[5];
    const float* dtproj_b = (const float*)d_in[6];
    const float* A_log    = (const float*)d_in[7];
    const float* Dp       = (const float*)d_in[8];
    const float* out_w    = (const float*)d_in[9];
    const float* norm_w   = (const float*)d_in[10];
    float* out = (float*)d_out;

    char* w = (char*)d_ws;
    size_t off = 0;
    auto alloc = [&](size_t bytes) -> void* {
        void* p = w + off;
        off = (off + bytes + 255) & ~(size_t)255;
        return p;
    };
    ushort_t* xn_bf   = (ushort_t*)alloc((size_t)ROWS * D_MODEL * 2);
    ushort_t* w1_bf   = (ushort_t*)alloc((size_t)2 * D_INNER * D_MODEL * 2);
    ushort_t* wx_bf   = (ushort_t*)alloc((size_t)96 * D_INNER * 2);
    ushort_t* wdt_bf  = (ushort_t*)alloc((size_t)D_INNER * DT_RANK * 2);
    ushort_t* wo_bf   = (ushort_t*)alloc((size_t)D_MODEL * D_INNER * 2);
    float*    xz      = (float*)alloc((size_t)ROWS * 2 * D_INNER * 4);
    float*    x_conv  = (float*)alloc((size_t)ROWS * D_INNER * 4);
    ushort_t* xconv_bf= (ushort_t*)alloc((size_t)ROWS * D_INNER * 2);
    float*    x_dbc   = (float*)alloc((size_t)ROWS * 96 * 4);
    ushort_t* dtraw_bf= (ushort_t*)alloc((size_t)ROWS * DT_RANK * 2);
    float*    dc      = (float*)alloc((size_t)ROWS * D_INNER * 4);  // pre -> dc
    float*    du      = (float*)alloc((size_t)ROWS * D_INNER * 4);
    ushort_t* yg_bf   = (ushort_t*)alloc((size_t)ROWS * D_INNER * 2);

    // 1) weight conversions to bf16
    {
        int n1 = 2 * D_INNER * D_MODEL;
        f2bf_kernel<<<(n1 + 255) / 256, 256, 0, stream>>>(in_w, w1_bf, n1);
        int n2 = 96 * D_INNER;
        f2bf_kernel<<<(n2 + 255) / 256, 256, 0, stream>>>(xproj_w, wx_bf, n2);
        int n3 = D_INNER * DT_RANK;
        f2bf_kernel<<<(n3 + 255) / 256, 256, 0, stream>>>(dtproj_w, wdt_bf, n3);
        int n4 = D_MODEL * D_INNER;
        f2bf_kernel<<<(n4 + 255) / 256, 256, 0, stream>>>(out_w, wo_bf, n4);
    }
    // 2) RMSNorm
    rmsnorm_kernel<<<ROWS, 256, 0, stream>>>(x, norm_w, xn_bf);
    // 3) in_proj: xz[4096,4096] = xn @ in_w^T
    {
        dim3 grid((2 * D_INNER) / BN, ROWS / BM);
        gemm_bf16_wmma<<<grid, 128, 0, stream>>>(xn_bf, w1_bf, xz, nullptr,
                                                 ROWS, 2 * D_INNER, D_MODEL, 2 * D_INNER);
    }
    // 4) depthwise conv + silu
    conv_silu_kernel<<<(ROWS * D_INNER) / 256, 256, 0, stream>>>(xz, conv_w, conv_b,
                                                                 x_conv, xconv_bf);
    // 5) x_proj: x_dbc[4096,96] = x_conv @ xproj_w^T
    {
        dim3 grid((96 + BN - 1) / BN, ROWS / BM);
        gemm_bf16_wmma<<<grid, 128, 0, stream>>>(xconv_bf, wx_bf, x_dbc, nullptr,
                                                 ROWS, 96, D_INNER, 96);
    }
    // 6) dt_raw -> bf16
    dtraw_bf_kernel<<<(ROWS * DT_RANK) / 256, 256, 0, stream>>>(x_dbc, dtraw_bf);
    // 7) dt_proj: pre[4096,2048] = dt_raw @ dtproj_w^T
    {
        dim3 grid(D_INNER / BN, ROWS / BM);
        gemm_bf16_wmma<<<grid, 128, 0, stream>>>(dtraw_bf, wdt_bf, dc, nullptr,
                                                 ROWS, D_INNER, DT_RANK, D_INNER);
    }
    // 8) softplus + clip + du
    dt_epilogue_kernel<<<(ROWS * D_INNER) / 256, 256, 0, stream>>>(dc, dtproj_b,
                                                                   x_conv, du);
    // 9) sequential SSM scan + D-skip + gating -> yg bf16
    scan_kernel<<<(BATCH * D_INNER) / 256, 256, 0, stream>>>(dc, du, x_dbc, x_conv,
                                                             xz, A_log, Dp, yg_bf);
    // 10) out_proj + residual: out[4096,1024] = yg @ out_w^T + x
    {
        dim3 grid(D_MODEL / BN, ROWS / BM);
        gemm_bf16_wmma<<<grid, 128, 0, stream>>>(yg_bf, wo_bf, out, x,
                                                 ROWS, D_MODEL, D_INNER, D_MODEL);
    }
    (void)in_sizes; (void)n_in; (void)out_size; (void)ws_size;
}
